// TextDecoder_14010183320221
// MI455X (gfx1250) — compile-verified, matrix operand order checked by reading, not run on hardware
//
#include <hip/hip_runtime.h>
#include <hip/hip_bf16.h>

// ---------------------------------------------------------------------------
// Persistent bf16-WMMA LSTM decoder for MI455X (gfx1250, wave32).
//   - weights packed once to bf16 WMMA-fragment layout (resident in L2)
//   - one persistent kernel runs all 512 steps; phases separated by an
//     atomic grid barrier (s_sleep spin on one lane per WG)
//   - A-panels staged into LDS with global_load_async_to_lds_b128 (ASYNCcnt)
//   - each wave computes a 16x64 macro-tile: 1 LDS A-frag feeds 4 WMMAs
//   - K-loop double-buffered (disjoint reg sets) so B loads overlap WMMA
// ---------------------------------------------------------------------------

typedef __attribute__((ext_vector_type(16))) __bf16 v16bf;
typedef __attribute__((ext_vector_type(8)))  float  v8f;

constexpr int B = 256, D = 128, H = 512, V = 256, T = 512;
constexpr int K1  = 2 * H;        // 1024 : A columns for LSTM GEMMs ([x|h])
constexpr int N1  = 4 * H;        // 2048 : gate columns
constexpr int MT  = B / 16;       // 16 M-tiles
constexpr int KT1 = K1 / 32;      // 32 K-slices (LSTM)
constexpr int NT1 = N1 / 16;      // 128 N-tiles (LSTM)
constexpr int KTP = H / 32;       // 16 K-slices (projection)
constexpr int NTP = V / 16;       // 16 N-tiles (projection)

constexpr int NWG = 64;           // persistent workgroups (16 mt x 4 n-quads)
constexpr int THREADS = 256;      // 8 waves per WG (wave32)
constexpr int WAVES = THREADS / 32;

// ---- workspace layout (bytes) ----
constexpr size_t OFF_CNT  = 0;                                   // barrier ctr
constexpr size_t OFF_WB1  = 256;
constexpr size_t SZ_WB1   = (size_t)NT1 * KT1 * 1024;            // 4 MB
constexpr size_t OFF_WB2  = OFF_WB1 + SZ_WB1;
constexpr size_t OFF_WBP  = OFF_WB2 + SZ_WB1;
constexpr size_t SZ_WBP   = (size_t)NTP * KTP * 1024;            // 256 KB
constexpr size_t OFF_APK1 = OFF_WBP + SZ_WBP;                    // [x|h1] packed
constexpr size_t SZ_APK   = (size_t)MT * KT1 * 1024;             // 512 KB
constexpr size_t OFF_APK2 = OFF_APK1 + SZ_APK;                   // [h1|h2] packed
constexpr size_t OFF_APKP = OFF_APK2 + SZ_APK;                   // h2 packed
constexpr size_t SZ_APKP  = (size_t)MT * KTP * 1024;             // 256 KB
constexpr size_t OFF_G1   = OFF_APKP + SZ_APKP;                  // gates1 f32
constexpr size_t SZ_G     = (size_t)B * N1 * 4;                  // 2 MB
constexpr size_t OFF_G2   = OFF_G1 + SZ_G;
constexpr size_t OFF_C1   = OFF_G2 + SZ_G;                       // cell states
constexpr size_t SZ_C     = (size_t)B * H * 4;                   // 512 KB
constexpr size_t OFF_C2   = OFF_C1 + SZ_C;

// ---------------------------------------------------------------------------
// helpers
// ---------------------------------------------------------------------------

// LDS byte-offset of a __shared__ object (generic -> local addrspacecast).
__device__ __forceinline__ unsigned ldsOffset(const void* p) {
    return (unsigned)(unsigned long long)
        (__attribute__((address_space(3))) const char*)p;
}

// Async copy 16B global -> LDS (per-lane addresses, ASYNCcnt-tracked).
__device__ __forceinline__ void asyncCopy16(unsigned ldsOff, const void* gsrc) {
    unsigned long long ga = (unsigned long long)gsrc;
    asm volatile("global_load_async_to_lds_b128 %0, %1, off"
                 :: "v"(ldsOff), "v"(ga) : "memory");
}
__device__ __forceinline__ void asyncWait0() {
    asm volatile("s_wait_asynccnt 0x0" ::: "memory");
}

// Write one scalar into packed WMMA A-fragment layout (16x32 bf16 frags).
// Fragment (mt,kt): lane l holds 16 bf16; element i maps to
//   k = kt*32 + (i>=8)*16 + (l>>4)*8 + (i&7),  m = mt*16 + (l&15)
__device__ __forceinline__ void storePacked(__bf16* apk, int ktTiles,
                                            int b, int k, float v) {
    int mt = b >> 4, m = b & 15;
    int kt = k >> 5, kk = k & 31;
    int h, i;
    if (kk < 16) { h = kk >> 3; i = kk & 7; }
    else         { int r = kk - 16; h = r >> 3; i = 8 + (r & 7); }
    int lane = (h << 4) | m;
    apk[(size_t)(mt * ktTiles + kt) * 512 + lane * 16 + i] = (__bf16)v;
}

// Monotonic device-wide barrier (counter zeroed at launch by k_init).
__device__ __forceinline__ void gridBarrier(unsigned* cnt, unsigned target) {
    __threadfence();
    __syncthreads();
    if (threadIdx.x == 0) {
        atomicAdd(cnt, 1u);
        while (__hip_atomic_load(cnt, __ATOMIC_ACQUIRE,
                                 __HIP_MEMORY_SCOPE_AGENT) < target)
            __builtin_amdgcn_s_sleep(1);
    }
    __syncthreads();
    __threadfence();
}

// One 16x16 tile straight from global packed operands (projection phase).
__device__ __forceinline__ v8f wmmaTile(const __bf16* apk, const __bf16* wpk,
                                        int mt, int nt, int ktTiles, int lane) {
    v8f acc = {0.f, 0.f, 0.f, 0.f, 0.f, 0.f, 0.f, 0.f};
    const __bf16* abase = apk + (size_t)mt * ktTiles * 512;
    const __bf16* bbase = wpk + (size_t)nt * ktTiles * 512;
    for (int kt = 0; kt < ktTiles; ++kt) {
        v16bf a = ((const v16bf*)(abase + (size_t)kt * 512))[lane];
        v16bf bfrag = ((const v16bf*)(bbase + (size_t)kt * 512))[lane];
        acc = __builtin_amdgcn_wmma_f32_16x16x32_bf16(
            false, a, false, bfrag, (short)0, acc, false, false);
    }
    return acc;
}

// ---------------------------------------------------------------------------
// setup kernels
// ---------------------------------------------------------------------------

__global__ void k_init(char* ws) {
    int idx = blockIdx.x * THREADS + threadIdx.x;
    if (idx == 0) *(unsigned*)(ws + OFF_CNT) = 0u;
    float* C1 = (float*)(ws + OFF_C1);
    float* C2 = (float*)(ws + OFF_C2);
    if (idx < B * H) { C1[idx] = 0.f; C2[idx] = 0.f; }
}

// Pack [Wih;Whh] (both layers) and Wp into bf16 WMMA B-fragment layout.
__global__ void k_pack(char* ws, const float* Wih1, const float* Whh1,
                       const float* Wih2, const float* Whh2, const float* Wp) {
    long idx = (long)blockIdx.x * THREADS + threadIdx.x;
    constexpr long L1 = (long)NT1 * KT1 * 512;
    constexpr long LP = (long)NTP * KTP * 512;
    __bf16* dst; const float* A; const float* Bm; int KT; long li = idx;
    if (idx < L1)            { dst = (__bf16*)(ws + OFF_WB1); A = Wih1; Bm = Whh1; KT = KT1; }
    else if (idx < 2 * L1)   { li -= L1;     dst = (__bf16*)(ws + OFF_WB2); A = Wih2; Bm = Whh2; KT = KT1; }
    else if (idx < 2*L1+LP)  { li -= 2 * L1; dst = (__bf16*)(ws + OFF_WBP); A = Wp;   Bm = nullptr; KT = KTP; }
    else return;
    long frag = li >> 9; int within = (int)(li & 511);
    int lane = within >> 4, i = within & 15;
    int nt = (int)(frag / KT), kt = (int)(frag % KT);
    int n = nt * 16 + (lane & 15);
    int k = kt * 32 + ((i >= 8) ? 16 : 0) + ((lane >> 4) << 3) + (i & 7);
    float v;
    if (Bm) v = (k < H) ? A[(size_t)n * H + k] : Bm[(size_t)n * H + (k - H)];
    else    v = A[(size_t)n * H + k];
    dst[li] = (__bf16)v;
}

// Expand MLP -> h1_0 = h2_0 = expanded; also seeds x0 = start_emb broadcast.
__global__ void k_expand(char* ws, const float* attract,
                         const float* W1, const float* b1,
                         const float* W2, const float* b2, const float* start) {
    __shared__ float xrow[D];
    __shared__ float hid[256];
    int b = blockIdx.x, t = threadIdx.x;
    if (t < D) xrow[t] = attract[(size_t)b * D + t];
    __syncthreads();
    float s = b1[t];
    for (int d = 0; d < D; ++d) s += W1[(size_t)t * D + d] * xrow[d];
    hid[t] = fmaxf(s, 0.f);
    __syncthreads();
    __bf16* APK1 = (__bf16*)(ws + OFF_APK1);
    __bf16* APK2 = (__bf16*)(ws + OFF_APK2);
    for (int q = 0; q < 2; ++q) {
        int j = t + q * 256;
        float s2 = b2[j];
        for (int k = 0; k < 256; ++k) s2 += W2[(size_t)j * 256 + k] * hid[k];
        float v = fmaxf(s2, 0.f);
        storePacked(APK1, KT1, b, H + j, v);   // h1_0
        storePacked(APK2, KT1, b, j,     v);   // layer2 x input = h1_0
        storePacked(APK2, KT1, b, H + j, v);   // h2_0
        storePacked(APK1, KT1, b, j, start[j]); // x0 broadcast
    }
}

// ---------------------------------------------------------------------------
// persistent decoder
// ---------------------------------------------------------------------------

__global__ __launch_bounds__(THREADS, 1)
void decoder_persistent(char* ws,
                        const float* bih1, const float* bhh1,
                        const float* bih2, const float* bhh2,
                        const float* bp, const float* Wp, float* out) {
    unsigned* cnt = (unsigned*)(ws + OFF_CNT);
    __bf16* WB1  = (__bf16*)(ws + OFF_WB1);
    __bf16* WB2  = (__bf16*)(ws + OFF_WB2);
    __bf16* WBP  = (__bf16*)(ws + OFF_WBP);
    __bf16* APK1 = (__bf16*)(ws + OFF_APK1);
    __bf16* APK2 = (__bf16*)(ws + OFF_APK2);
    __bf16* APKP = (__bf16*)(ws + OFF_APKP);
    float*  G1   = (float*)(ws + OFF_G1);
    float*  G2   = (float*)(ws + OFF_G2);
    float*  C1   = (float*)(ws + OFF_C1);
    float*  C2   = (float*)(ws + OFF_C2);

    __shared__ __align__(16) char sA[KT1 * 1024];  // 32 KB A-panel
    __shared__ float sLog[16][V + 1];              // projection logits

    const int lane = threadIdx.x & 31;
    const int wid  = threadIdx.x >> 5;
    const int mtWG = blockIdx.x >> 2;              // this WG's M-tile
    const int ntBase = (blockIdx.x & 3) * 32;      // this WG's 32-wide N slice
    unsigned epoch = 0;

    for (int t = 0; t < T; ++t) {
        // ================= Phase A/C: the two LSTM-gate GEMMs ==============
        #pragma unroll 1
        for (int layer = 0; layer < 2; ++layer) {
            const __bf16* apk = layer ? APK2 : APK1;
            const __bf16* wpk = layer ? WB2 : WB1;
            float* G = layer ? G2 : G1;

            // stage A panel (mt, all 32 K-slices) into LDS asynchronously
            {
                const char* src = (const char*)(apk + (size_t)mtWG * KT1 * 512);
                unsigned ldsBase = ldsOffset(sA);
                for (int c = threadIdx.x; c < (KT1 * 1024) / 16; c += THREADS)
                    asyncCopy16(ldsBase + c * 16, src + c * 16);
                asyncWait0();
            }
            __syncthreads();

            // each wave: 16x64 macro-tile (4 acc share one LDS A fragment).
            // K-loop double-buffered: loads for slice kt+1 overlap the 4
            // WMMAs of slice kt (disjoint register sets aA/bA vs aB/bB).
            const int nt0 = ntBase + wid * 4;
            const __bf16* bb = wpk + (size_t)nt0 * KT1 * 512;

            v8f acc[4];
            #pragma unroll
            for (int q = 0; q < 4; ++q)
                acc[q] = (v8f){0.f, 0.f, 0.f, 0.f, 0.f, 0.f, 0.f, 0.f};

            auto loadB4 = [&](v16bf bf[4], int kt) {
                #pragma unroll
                for (int q = 0; q < 4; ++q)
                    bf[q] = ((const v16bf*)(bb + (size_t)(q * KT1 + kt) * 512))[lane];
            };
            auto loadA = [&](int kt) -> v16bf {
                return *(const v16bf*)(sA + kt * 1024 + lane * 32);
            };

            v16bf bA[4], bB[4], aA, aB;
            loadB4(bA, 0);
            aA = loadA(0);
            #pragma unroll 2
            for (int kt = 0; kt < KT1; kt += 2) {
                // prefill odd slice while even slice computes
                loadB4(bB, kt + 1);
                aB = loadA(kt + 1);
                if (kt + 4 < KT1)
                    __builtin_prefetch(bb + (size_t)(kt + 4) * 512 + lane * 16, 0, 1);
                #pragma unroll
                for (int q = 0; q < 4; ++q)
                    acc[q] = __builtin_amdgcn_wmma_f32_16x16x32_bf16(
                        false, aA, false, bA[q], (short)0, acc[q], false, false);
                // prefill next even slice while odd slice computes
                if (kt + 2 < KT1) {
                    loadB4(bA, kt + 2);
                    aA = loadA(kt + 2);
                }
                #pragma unroll
                for (int q = 0; q < 4; ++q)
                    acc[q] = __builtin_amdgcn_wmma_f32_16x16x32_bf16(
                        false, aB, false, bB[q], (short)0, acc[q], false, false);
            }

            int mb = mtWG * 16 + ((lane >> 4) << 3);
            #pragma unroll
            for (int q = 0; q < 4; ++q) {
                int n = (nt0 + q) * 16 + (lane & 15);
                #pragma unroll
                for (int r = 0; r < 8; ++r)
                    G[(size_t)(mb + r) * N1 + n] = acc[q][r];
            }
            gridBarrier(cnt, ++epoch * NWG);

            // ---- Phase B/D: LSTM cell elementwise update ----
            const float* bih = layer ? bih2 : bih1;
            const float* bhh = layer ? bhh2 : bhh1;
            float* C = layer ? C2 : C1;
            for (int e = blockIdx.x * THREADS + threadIdx.x; e < B * H;
                 e += NWG * THREADS) {
                int b = e >> 9, j = e & 511;
                const float* g = G + (size_t)b * N1;
                float gi = g[j]       + bih[j]       + bhh[j];
                float gf = g[j + H]   + bih[j + H]   + bhh[j + H];
                float gg = g[j + 2*H] + bih[j + 2*H] + bhh[j + 2*H];
                float go = g[j + 3*H] + bih[j + 3*H] + bhh[j + 3*H];
                float si = 1.f / (1.f + __expf(-gi));
                float sf = 1.f / (1.f + __expf(-gf));
                float tg = tanhf(gg);
                float so = 1.f / (1.f + __expf(-go));
                float c  = sf * C[e] + si * tg;
                C[e] = c;
                float h = so * tanhf(c);
                if (layer == 0) {
                    storePacked(APK1, KT1, b, H + j, h);  // next-step h1
                    storePacked(APK2, KT1, b, j,     h);  // layer2 x input
                } else {
                    storePacked(APK2, KT1, b, H + j, h);  // next-step h2
                    storePacked(APKP, KTP, b, j,     h);  // projection input
                }
            }
            gridBarrier(cnt, ++epoch * NWG);
        }

        // ====== Phase E (WGs 0..15): projection + argmax + feedback ========
        if (blockIdx.x < MT) {
            int mt = blockIdx.x;
            for (int q = 0; q < 2; ++q) {              // 16 N-tiles over 8 waves
                int nt = wid * 2 + q;
                v8f acc = wmmaTile(APKP, WBP, mt, nt, KTP, lane);
                int n  = nt * 16 + (lane & 15);
                int mb = (lane >> 4) << 3;             // local row base
                #pragma unroll
                for (int r = 0; r < 8; ++r) {
                    float v = acc[r] + bp[n];
                    sLog[mb + r][n] = v;
                    out[((size_t)(mt * 16 + mb + r) * T + t) * V + n] = v;
                }
            }
            __syncthreads();
            for (int q = 0; q < 2; ++q) {              // argmax: 2 rows per wave
                int lr = wid * 2 + q;
                int b  = mt * 16 + lr;
                float mx = -__builtin_inff(); int ix = 0;
                for (int n = lane; n < V; n += 32) {
                    float v = sLog[lr][n];
                    if (v > mx) { mx = v; ix = n; }
                }
                for (int off = 16; off; off >>= 1) {   // first-max tie-break
                    float om = __shfl_xor(mx, off, 32);
                    int   oi = __shfl_xor(ix, off, 32);
                    if (om > mx || (om == mx && oi < ix)) { mx = om; ix = oi; }
                }
                const float* wrow = Wp + (size_t)ix * H;   // x_next = Wp[pred]
                for (int j = lane; j < H; j += 32)
                    storePacked(APK1, KT1, b, j, wrow[j]);
            }
        }
        gridBarrier(cnt, ++epoch * NWG);
    }
}

// ---------------------------------------------------------------------------
// entry
// ---------------------------------------------------------------------------

extern "C" void kernel_launch(void* const* d_in, const int* in_sizes, int n_in,
                              void* d_out, int out_size, void* d_ws, size_t ws_size,
                              hipStream_t stream) {
    const float* attract = (const float*)d_in[0];
    const float* W1   = (const float*)d_in[1];
    const float* b1   = (const float*)d_in[2];
    const float* W2   = (const float*)d_in[3];
    const float* b2   = (const float*)d_in[4];
    const float* Wih1 = (const float*)d_in[5];
    const float* Whh1 = (const float*)d_in[6];
    const float* bih1 = (const float*)d_in[7];
    const float* bhh1 = (const float*)d_in[8];
    const float* Wih2 = (const float*)d_in[9];
    const float* Whh2 = (const float*)d_in[10];
    const float* bih2 = (const float*)d_in[11];
    const float* bhh2 = (const float*)d_in[12];
    const float* Wp   = (const float*)d_in[13];
    const float* bp   = (const float*)d_in[14];
    const float* start = (const float*)d_in[15];
    char*  ws  = (char*)d_ws;
    float* out = (float*)d_out;

    k_init<<<(B * H + THREADS - 1) / THREADS, THREADS, 0, stream>>>(ws);

    long packTotal = 2L * NT1 * KT1 * 512 + (long)NTP * KTP * 512;
    k_pack<<<(int)((packTotal + THREADS - 1) / THREADS), THREADS, 0, stream>>>(
        ws, Wih1, Whh1, Wih2, Whh2, Wp);

    k_expand<<<B, THREADS, 0, stream>>>(ws, attract, W1, b1, W2, b2, start);

    decoder_persistent<<<NWG, THREADS, 0, stream>>>(
        ws, bih1, bhh1, bih2, bhh2, bp, Wp, out);
}
